// CausalGroupQueryAttention_28819230556254
// MI455X (gfx1250) — matrix operand order repaired and checked by reading, not compiled, not run on hardware
//
#include <hip/hip_runtime.h>
#include <hip/hip_bf16.h>
#include <math.h>

// ---------------- problem constants ----------------
#define HIDDEN   2048
#define NUM_HEADS 16
#define NUM_KV_HEADS 4
#define HEAD_DIM 128
#define N_REP    4
#define KV_DIM   512
#define BATCH    2
#define SEQ      2048
#define MTOT     (BATCH * SEQ)   // 4096 rows

// ---------------- WMMA fragment types ----------------
typedef __attribute__((ext_vector_type(16))) __bf16         v16bf;   // 32 B A/B fragment
typedef __attribute__((ext_vector_type(8)))  float          v8f;     // C/D fragment
typedef __attribute__((ext_vector_type(8)))  unsigned short ushort8; // 16 B chunk
typedef int v4i __attribute__((vector_size(16)));                    // 16 B async-copy unit

union Frag16 {
  v16bf   v;
  ushort8 h[2];
};

__device__ __forceinline__ v8f v8f_zero() {
  v8f z;
#pragma unroll
  for (int i = 0; i < 8; ++i) z[i] = 0.0f;
  return z;
}

__device__ __forceinline__ unsigned short f32_to_bf16(float f) {
  union { float f; unsigned int u; } x;
  x.f = f;
  unsigned int u = x.u;
  u += 0x7FFFu + ((u >> 16) & 1u);   // round-to-nearest-even
  return (unsigned short)(u >> 16);
}

// -------- CDNA5 async global->LDS copy (ASYNCcnt path), with sync fallback --------
#if defined(__HIP_DEVICE_COMPILE__) && __has_builtin(__builtin_amdgcn_global_load_async_to_lds_b128)
#define HAVE_ASYNC_LDS 1
#else
#define HAVE_ASYNC_LDS 0
#endif

__device__ __forceinline__ void cp16(const unsigned short* g, unsigned short* l) {
#if HAVE_ASYNC_LDS
  __builtin_amdgcn_global_load_async_to_lds_b128(
      (__attribute__((address_space(1))) v4i*)g,
      (__attribute__((address_space(3))) v4i*)l, 0, 0);
#else
  *(ushort8*)l = *(const ushort8*)g;
#endif
}

__device__ __forceinline__ void async_fence() {
#if HAVE_ASYNC_LDS
  asm volatile("s_wait_asynccnt 0x0" ::: "memory");
#endif
}

// ---------------- conversion kernels ----------------
__global__ void cvt_f32_bf16(const float* __restrict__ in,
                             unsigned short* __restrict__ out, int n) {
  int i = blockIdx.x * blockDim.x + threadIdx.x;
  if (i < n) out[i] = f32_to_bf16(in[i]);
}

// WT[n*K + k] = bf16(W[k*N + n])   (W is [K,N] fp32, WT is [N,K] bf16)
__global__ void transpose_cvt(const float* __restrict__ W,
                              unsigned short* __restrict__ WT, int K, int N) {
  int i = blockIdx.x * blockDim.x + threadIdx.x;
  if (i >= N * K) return;
  int n = i / K, k = i % K;
  WT[i] = f32_to_bf16(W[(size_t)k * N + n]);
}

// ---------------- cooperative WMMA GEMM ----------------
// Block: 256 threads (8 waves). Macro-tile: 128(M) x 64(N), K-step 32.
// A: [M,K] bf16 row-major. WT: [N,K] bf16 (transposed weights).
// A & B k-slices double-buffered in LDS via async loads.
// mode 0: bf16 out row-major [M,N]
// mode 1: bf16 out as K-cache [B,HKV,T,128]
// mode 2: bf16 out as V^T     [B,HKV,128,T]
// mode 3: f32  out row-major [M,N]
#define A_STRIDE 40   // 32 data shorts + 8 pad (80 B rows -> conflict-free frag reads)
#define B_STRIDE 40

__global__ __launch_bounds__(256)
void gemm_wmma(const unsigned short* __restrict__ A,
               const unsigned short* __restrict__ WT,
               const float* __restrict__ bias,
               void* __restrict__ out,
               int M, int N, int K, int mode) {
  __shared__ __align__(16) unsigned short Abuf[2][128 * A_STRIDE];
  __shared__ __align__(16) unsigned short Bbuf[2][64 * B_STRIDE];

  const int tid  = threadIdx.x;
  const int lane = tid & 31;
  const int w    = tid >> 5;       // wave id 0..7 -> M stripe
  const int l16  = lane & 15;
  const int g    = lane >> 4;

  const int tiles_n = N >> 6;
  const int tm = blockIdx.x / tiles_n;
  const int tn = blockIdx.x % tiles_n;
  const int m0 = tm << 7;          // 128-row macro tile
  const int n0 = tn << 6;          // 64-col macro tile

  // ---- staging: 3 x 16B chunks per thread per k-step ----
  const int ar0 = tid >> 2,          ap0 = tid & 3;          // A chunk 0
  const int ar1 = (tid + 256) >> 2,  ap1 = tid & 3;          // A chunk 1
  const int br  = tid >> 2,          bp  = tid & 3;          // B chunk

  auto stage = [&](int kk, int buf) {
    cp16(A  + (size_t)(m0 + ar0) * K + kk + ap0 * 8, &Abuf[buf][ar0 * A_STRIDE + ap0 * 8]);
    cp16(A  + (size_t)(m0 + ar1) * K + kk + ap1 * 8, &Abuf[buf][ar1 * A_STRIDE + ap1 * 8]);
    cp16(WT + (size_t)(n0 + br ) * K + kk + bp  * 8, &Bbuf[buf][br  * B_STRIDE + bp  * 8]);
  };

  v8f acc[4];
#pragma unroll
  for (int t = 0; t < 4; ++t) acc[t] = v8f_zero();

  const int NK = K >> 5;          // k-steps of 32
  stage(0, 0);
  async_fence();
  __syncthreads();

  for (int ks = 0; ks < NK; ++ks) {
    const int cur = ks & 1;
    if (ks + 1 < NK) stage((ks + 1) << 5, cur ^ 1);

    const unsigned short* As = &Abuf[cur][0];
    const unsigned short* Bs = &Bbuf[cur][0];

    // ---- load ALL fragments first (distinct registers -> one LDS drain) ----
    Frag16 af;
    af.h[0] = *(const ushort8*)(As + (w * 16 + l16) * A_STRIDE + g * 8);
    af.h[1] = *(const ushort8*)(As + (w * 16 + l16) * A_STRIDE + 16 + g * 8);
    Frag16 bfg[4];
#pragma unroll
    for (int t = 0; t < 4; ++t) {
      const unsigned short* brow = Bs + (t * 16 + l16) * B_STRIDE + g * 8;
      bfg[t].h[0] = *(const ushort8*)(brow);
      bfg[t].h[1] = *(const ushort8*)(brow + 16);
    }
    // ---- then fire 4 back-to-back WMMAs (no RAW between them) ----
#pragma unroll
    for (int t = 0; t < 4; ++t)
      acc[t] = __builtin_amdgcn_wmma_f32_16x16x32_bf16(
          false, af.v, false, bfg[t].v, (short)0, acc[t], false, false);

    async_fence();        // next-buffer async loads complete
    __syncthreads();      // all waves done reading cur / writing next
  }

  // ---- epilogue ----
#pragma unroll
  for (int t = 0; t < 4; ++t) {
    const int n = n0 + t * 16 + l16;
    const float bv = bias ? bias[n] : 0.0f;
#pragma unroll
    for (int r = 0; r < 8; ++r) {
      const int m = m0 + w * 16 + r + 8 * g;
      const float val = acc[t][r] + bv;
      if (mode == 3) {
        ((float*)out)[(size_t)m * N + n] = val;
      } else {
        unsigned short hv = f32_to_bf16(val);
        size_t idx;
        if (mode == 0) {
          idx = (size_t)m * N + n;
        } else {
          const int b  = m / SEQ;
          const int tt = m % SEQ;
          const int hk = n >> 7;
          const int d  = n & 127;
          if (mode == 1)
            idx = (((size_t)(b * NUM_KV_HEADS + hk)) * SEQ + tt) * HEAD_DIM + d;
          else
            idx = (((size_t)(b * NUM_KV_HEADS + hk)) * HEAD_DIM + d) * SEQ + tt;
        }
        ((unsigned short*)out)[idx] = hv;
      }
    }
  }
}

// ---------------- causal flash attention, one wave per 16-row tile per head ----------------
// Q : [B, T, H*128] bf16 row-major
// Kb: [B, HKV, T, 128] bf16
// Vt: [B, HKV, 128, T] bf16 (V transposed)
// O : [B, T, H*128] bf16 row-major
__global__ void flash_attn(const unsigned short* __restrict__ Q,
                           const unsigned short* __restrict__ Kb,
                           const unsigned short* __restrict__ Vt,
                           unsigned short* __restrict__ O) {
  __shared__ __align__(16) unsigned short Pl[16 * 32];

  const int lane = threadIdx.x;
  const int l16  = lane & 15;
  const int g    = lane >> 4;
  const int tile = blockIdx.x;
  const int h    = blockIdx.y;
  const int b    = blockIdx.z;
  const int hkv  = h / N_REP;
  const int i0   = tile * 16;
  const float scale = 0.08838834764831845f; // 1/sqrt(128)

  const unsigned short* qrow =
      Q + ((size_t)(b * SEQ + i0 + l16)) * HIDDEN + h * HEAD_DIM;
  Frag16 aq[4];
#pragma unroll
  for (int c = 0; c < 4; ++c) {
    aq[c].h[0] = *(const ushort8*)(qrow + c * 32 + g * 8);
    aq[c].h[1] = *(const ushort8*)(qrow + c * 32 + 16 + g * 8);
  }

  const unsigned short* kbase =
      Kb + ((size_t)(b * NUM_KV_HEADS + hkv)) * SEQ * HEAD_DIM;
  const unsigned short* vbase =
      Vt + ((size_t)(b * NUM_KV_HEADS + hkv)) * HEAD_DIM * SEQ;

  v8f oacc[8];
#pragma unroll
  for (int t = 0; t < 8; ++t) oacc[t] = v8f_zero();
  float mrun[8], lrun[8];
#pragma unroll
  for (int r = 0; r < 8; ++r) { mrun[r] = -3.0e38f; lrun[r] = 0.0f; }

  const int nsteps = (i0 + 15) / 32 + 1;
  for (int js = 0; js < nsteps; ++js) {
    const int s0 = js * 32;

    // prefetch next K tile rows (lowers to global_prefetch_b8)
    if (s0 + 32 + 31 < SEQ)
      __builtin_prefetch(kbase + (size_t)(s0 + 32 + lane) * HEAD_DIM, 0, 3);

    // ---- S = Q K^T over two 16-column tiles ----
    v8f sacc[2];
    sacc[0] = v8f_zero();
    sacc[1] = v8f_zero();
#pragma unroll
    for (int c = 0; c < 4; ++c) {
      Frag16 bk0, bk1;
      {
        const unsigned short* krow0 =
            kbase + (size_t)(s0 + l16) * HEAD_DIM + c * 32 + g * 8;
        const unsigned short* krow1 =
            kbase + (size_t)(s0 + 16 + l16) * HEAD_DIM + c * 32 + g * 8;
        bk0.h[0] = ((const ushort8*)krow0)[0];
        bk0.h[1] = ((const ushort8*)krow0)[2];
        bk1.h[0] = ((const ushort8*)krow1)[0];
        bk1.h[1] = ((const ushort8*)krow1)[2];
      }
      sacc[0] = __builtin_amdgcn_wmma_f32_16x16x32_bf16(
          false, aq[c].v, false, bk0.v, (short)0, sacc[0], false, false);
      sacc[1] = __builtin_amdgcn_wmma_f32_16x16x32_bf16(
          false, aq[c].v, false, bk1.v, (short)0, sacc[1], false, false);
    }

    // ---- scale + causal mask + row-max (row = r + 8*g across 16-lane halves) ----
    float rm[8];
#pragma unroll
    for (int r = 0; r < 8; ++r) {
      const int row = i0 + r + 8 * g;
      float s0v = sacc[0][r] * scale;
      float s1v = sacc[1][r] * scale;
      if (s0 + l16 > row)      s0v = -3.0e38f;
      if (s0 + 16 + l16 > row) s1v = -3.0e38f;
      sacc[0][r] = s0v;
      sacc[1][r] = s1v;
      float v = fmaxf(s0v, s1v);
      v = fmaxf(v, __shfl_xor(v, 1, 32));
      v = fmaxf(v, __shfl_xor(v, 2, 32));
      v = fmaxf(v, __shfl_xor(v, 4, 32));
      v = fmaxf(v, __shfl_xor(v, 8, 32));
      rm[r] = v;
    }

    // ---- online softmax update; P tile -> LDS (bf16) ----
    float alpha[8];
#pragma unroll
    for (int r = 0; r < 8; ++r) {
      const float mnew = fmaxf(mrun[r], rm[r]);
      alpha[r] = __expf(mrun[r] - mnew);
      mrun[r] = mnew;
      const float p0 = __expf(sacc[0][r] - mnew);
      const float p1 = __expf(sacc[1][r] - mnew);
      const int prow = r + 8 * g;
      Pl[prow * 32 + l16]      = f32_to_bf16(p0);
      Pl[prow * 32 + 16 + l16] = f32_to_bf16(p1);
      float ps = p0 + p1;
      ps += __shfl_xor(ps, 1, 32);
      ps += __shfl_xor(ps, 2, 32);
      ps += __shfl_xor(ps, 4, 32);
      ps += __shfl_xor(ps, 8, 32);
      lrun[r] = lrun[r] * alpha[r] + ps;
    }
#pragma unroll
    for (int t = 0; t < 8; ++t)
#pragma unroll
      for (int r = 0; r < 8; ++r) oacc[t][r] *= alpha[r];

    __syncthreads();
    Frag16 ap;   // C-layout -> A-fragment layout via LDS
    ap.h[0] = *(const ushort8*)(&Pl[l16 * 32 + g * 8]);
    ap.h[1] = *(const ushort8*)(&Pl[l16 * 32 + 16 + g * 8]);
    __syncthreads();

    // ---- O += P V : V^T rows give contiguous B fragments ----
#pragma unroll
    for (int t = 0; t < 8; ++t) {
      const unsigned short* vrow =
          vbase + (size_t)(t * 16 + l16) * SEQ + s0 + g * 8;
      Frag16 bv;
      bv.h[0] = ((const ushort8*)vrow)[0];
      bv.h[1] = ((const ushort8*)vrow)[2];
      oacc[t] = __builtin_amdgcn_wmma_f32_16x16x32_bf16(
          false, ap.v, false, bv.v, (short)0, oacc[t], false, false);
    }
  }

  // ---- normalize and store O ----
#pragma unroll
  for (int r = 0; r < 8; ++r) {
    const float inv = 1.0f / lrun[r];
    const int row = i0 + r + 8 * g;
    unsigned short* orow =
        O + ((size_t)(b * SEQ + row)) * HIDDEN + h * HEAD_DIM;
#pragma unroll
    for (int t = 0; t < 8; ++t)
      orow[t * 16 + l16] = f32_to_bf16(oacc[t][r] * inv);
  }
}

// ---------------- host-side launcher ----------------
extern "C" void kernel_launch(void* const* d_in, const int* in_sizes, int n_in,
                              void* d_out, int out_size, void* d_ws, size_t ws_size,
                              hipStream_t stream) {
  (void)in_sizes; (void)n_in; (void)out_size; (void)ws_size;

  const float* x  = (const float*)d_in[0];
  const float* wq = (const float*)d_in[1];
  const float* bq = (const float*)d_in[2];
  const float* wk = (const float*)d_in[3];
  const float* bk = (const float*)d_in[4];
  const float* wv = (const float*)d_in[5];
  const float* bv = (const float*)d_in[6];
  const float* wo = (const float*)d_in[7];
  float* out = (float*)d_out;

  char* ws = (char*)d_ws;
  size_t off = 0;
  auto alloc = [&](size_t bytes) -> char* {
    char* p = ws + off;
    off += (bytes + 255) & ~(size_t)255;
    return p;
  };
  unsigned short* xb  = (unsigned short*)alloc((size_t)MTOT * HIDDEN * 2);
  unsigned short* wqT = (unsigned short*)alloc((size_t)HIDDEN * HIDDEN * 2);
  unsigned short* wkT = (unsigned short*)alloc((size_t)KV_DIM * HIDDEN * 2);
  unsigned short* wvT = (unsigned short*)alloc((size_t)KV_DIM * HIDDEN * 2);
  unsigned short* woT = (unsigned short*)alloc((size_t)HIDDEN * HIDDEN * 2);
  unsigned short* Qb  = (unsigned short*)alloc((size_t)MTOT * HIDDEN * 2);
  unsigned short* Kbf = (unsigned short*)alloc((size_t)BATCH * NUM_KV_HEADS * SEQ * HEAD_DIM * 2);
  unsigned short* Vtb = (unsigned short*)alloc((size_t)BATCH * NUM_KV_HEADS * SEQ * HEAD_DIM * 2);
  unsigned short* Ab  = (unsigned short*)alloc((size_t)MTOT * HIDDEN * 2);

  const int nx = MTOT * HIDDEN;
  cvt_f32_bf16<<<(nx + 255) / 256, 256, 0, stream>>>(x, xb, nx);
  transpose_cvt<<<(HIDDEN * HIDDEN + 255) / 256, 256, 0, stream>>>(wq, wqT, HIDDEN, HIDDEN);
  transpose_cvt<<<(HIDDEN * KV_DIM + 255) / 256, 256, 0, stream>>>(wk, wkT, HIDDEN, KV_DIM);
  transpose_cvt<<<(HIDDEN * KV_DIM + 255) / 256, 256, 0, stream>>>(wv, wvT, HIDDEN, KV_DIM);
  transpose_cvt<<<(HIDDEN * HIDDEN + 255) / 256, 256, 0, stream>>>(wo, woT, HIDDEN, HIDDEN);

  // projections (cooperative 128x64 macro-tile GEMMs)
  gemm_wmma<<<(MTOT / 128) * (HIDDEN / 64), 256, 0, stream>>>(
      xb, wqT, bq, Qb, MTOT, HIDDEN, HIDDEN, 0);
  gemm_wmma<<<(MTOT / 128) * (KV_DIM / 64), 256, 0, stream>>>(
      xb, wkT, bk, Kbf, MTOT, KV_DIM, HIDDEN, 1);
  gemm_wmma<<<(MTOT / 128) * (KV_DIM / 64), 256, 0, stream>>>(
      xb, wvT, bv, Vtb, MTOT, KV_DIM, HIDDEN, 2);

  // attention
  dim3 agrid(SEQ / 16, NUM_HEADS, BATCH);
  flash_attn<<<agrid, 32, 0, stream>>>(Qb, Kbf, Vtb, Ab);

  // output projection (fp32 result)
  gemm_wmma<<<(MTOT / 128) * (HIDDEN / 64), 256, 0, stream>>>(
      Ab, woT, nullptr, out, MTOT, HIDDEN, HIDDEN, 3);
}
